// BiRGAT_7249904796307
// MI455X (gfx1250) — compile-verified, matrix operand order checked by reading, not run on hardware
//
#include <hip/hip_runtime.h>
#include <hip/hip_bf16.h>

// ---------------------------------------------------------------------------
// BiRGAT (2-layer bipartite GATv2) for MI455X / gfx1250.
//   GEMMs  : f16 WMMA (v_wmma_f32_16x16x32_f16), fp32 accumulate.
//            Weights pre-transposed to [N,K] so A and B fragments are both
//            two contiguous global_load_b128 per k-step; 32x32 tile per wave
//            (4 WMMAs per k-step, 2x2 fragment reuse).
//   Edges  : fp32 gather / segment-softmax (atomic max via ordered-uint) /
//            weighted scatter-add.  One wave32 per edge, 8 lanes per head.
// ---------------------------------------------------------------------------

#define NS   4096
#define NG   20000
#define NE   131072
#define DIN  256
#define NH   4
#define C1   64
#define C3   128
#define HC1  256          // NH*C1
#define HC3  512          // NH*C3
#define NEG_ENC 0x00800000u   // ordered-uint encoding of -FLT_MAX

typedef __attribute__((ext_vector_type(16))) _Float16 v16h;
typedef __attribute__((ext_vector_type(8)))  _Float16 v8h;
typedef __attribute__((ext_vector_type(8)))  float    v8f;

static inline int cdiv(int a, int b) { return (a + b - 1) / b; }

__device__ __forceinline__ float eluf(float x) { return x > 0.f ? x : (expf(x) - 1.f); }

// ---------------------------------------------------------------------------
// elementwise helpers
// ---------------------------------------------------------------------------
__global__ void k_f32_to_f16(const float* __restrict__ in, _Float16* __restrict__ out, int n) {
    int i = blockIdx.x * blockDim.x + threadIdx.x;
    if (i < n) out[i] = (_Float16)in[i];
}

// convert + transpose: in = W[K,N] f32 (row-major)  ->  out = Wt[N,K] f16
__global__ void k_f32_to_f16_t(const float* __restrict__ in, _Float16* __restrict__ out,
                               int K, int N) {
    int i = blockIdx.x * blockDim.x + threadIdx.x;   // i indexes Wt flat [N*K]
    if (i >= N * K) return;
    int n = i / K, k = i - n * K;
    out[i] = (_Float16)in[(size_t)k * N + n];
}

__global__ void k_fill_f32(float* __restrict__ p, float v, int n) {
    int i = blockIdx.x * blockDim.x + threadIdx.x;
    if (i < n) p[i] = v;
}

__global__ void k_fill_u32(unsigned* __restrict__ p, unsigned v, int n) {
    int i = blockIdx.x * blockDim.x + threadIdx.x;
    if (i < n) p[i] = v;
}

__global__ void k_rowbias(float* __restrict__ out, const float* __restrict__ bias,
                          int rows, int cols) {
    int i = blockIdx.x * blockDim.x + threadIdx.x;
    if (i < rows * cols) out[i] = bias[i % cols];
}

// elu in place + f16 snapshot (x1_gene)
__global__ void k_elu_f16copy(float* __restrict__ x, _Float16* __restrict__ xh, int n) {
    int i = blockIdx.x * blockDim.x + threadIdx.x;
    if (i < n) { float v = eluf(x[i]); x[i] = v; xh[i] = (_Float16)v; }
}

// x1_sample = elu(attn_out + tile(sl1, NH)) ; also f16 snapshot.  n = NS*HC1
__global__ void k_sample1(float* __restrict__ x, const float* __restrict__ sl,
                          _Float16* __restrict__ xh, int n) {
    int i = blockIdx.x * blockDim.x + threadIdx.x;
    if (i >= n) return;
    int col = i & (HC1 - 1);
    int s   = i >> 8;                      // HC1 == 256
    float v = eluf(x[i] + sl[(size_t)s * C1 + (col & (C1 - 1))]);
    x[i] = v; xh[i] = (_Float16)v;
}

// out = elu(mean_heads(acc3) + bias3 + sl3) ; n = NS*C3
__global__ void k_final(const float* __restrict__ acc3, const float* __restrict__ sl3,
                        const float* __restrict__ bias3, float* __restrict__ out, int n) {
    int i = blockIdx.x * blockDim.x + threadIdx.x;
    if (i >= n) return;
    int c = i & (C3 - 1);
    int s = i >> 7;                        // C3 == 128
    const float* a = acc3 + (size_t)s * HC3;
    float mean = 0.25f * (a[c] + a[C3 + c] + a[2 * C3 + c] + a[3 * C3 + c]);
    out[i] = eluf(mean + bias3[c] + sl3[i]);
}

// ---------------------------------------------------------------------------
// WMMA GEMM:  Y[M,N] = X[M,K](f16) @ W[K,N](f16) + bias[N],  fp32 out.
// W supplied TRANSPOSED as Wt[N,K].  256 threads = 8 waves in a 4x2 grid;
// each wave computes a 32x32 tile (2x2 sub-tiles, 4 WMMA per k-step).
// Requires K % 32 == 0 and N % 64 == 0 (true for all call sites).
// Fragment layout per CDNA5 ISA 7.12.2: lane L (row/col = L&15) holds halves
// [kb..kb+7] and [kb+16..kb+23], kb = 8*(L>=16) — contiguous in X/Wt rows.
// ---------------------------------------------------------------------------
__device__ __forceinline__ v16h ld_frag(const _Float16* p) {
    v8h lo = *(const v8h*)(p);
    v8h hi = *(const v8h*)(p + 16);
    v16h r;
#pragma unroll
    for (int j = 0; j < 8; ++j) { r[j] = lo[j]; r[8 + j] = hi[j]; }
    return r;
}

__global__ void __launch_bounds__(256)
k_gemm_wmma(const _Float16* __restrict__ X, const _Float16* __restrict__ Wt,
            const float* __restrict__ bias, float* __restrict__ Y,
            int M, int N, int K) {
    int wave = threadIdx.x >> 5;
    int lane = threadIdx.x & 31;
    int wm = wave & 3, wn = wave >> 2;                 // 4x2 wave grid
    int m0 = blockIdx.x * 128 + wm * 32;
    if (m0 >= M) return;                               // wave-uniform exit
    int n0 = blockIdx.y * 64 + wn * 32;
    int lr = lane & 15;
    int kb = (lane >> 4) * 8;                          // K sub-block base: 0 or 8

    int ra0 = m0 + lr;      if (ra0 > M - 1) ra0 = M - 1;   // clamp (store-masked)
    int ra1 = m0 + 16 + lr; if (ra1 > M - 1) ra1 = M - 1;
    int cb0 = n0 + lr;                                 // N % 64 == 0 -> in range
    int cb1 = cb0 + 16;

    const _Float16* pA0 = X  + (size_t)ra0 * K + kb;
    const _Float16* pA1 = X  + (size_t)ra1 * K + kb;
    const _Float16* pB0 = Wt + (size_t)cb0 * K + kb;
    const _Float16* pB1 = Wt + (size_t)cb1 * K + kb;

    v8f acc00 = {}, acc01 = {}, acc10 = {}, acc11 = {};
#pragma unroll 4
    for (int k0 = 0; k0 < K; k0 += 32) {
        v16h a0 = ld_frag(pA0 + k0);
        v16h a1 = ld_frag(pA1 + k0);
        v16h b0 = ld_frag(pB0 + k0);
        v16h b1 = ld_frag(pB1 + k0);
        acc00 = __builtin_amdgcn_wmma_f32_16x16x32_f16(false, a0, false, b0, (short)0, acc00, false, false);
        acc01 = __builtin_amdgcn_wmma_f32_16x16x32_f16(false, a0, false, b1, (short)0, acc01, false, false);
        acc10 = __builtin_amdgcn_wmma_f32_16x16x32_f16(false, a1, false, b0, (short)0, acc10, false, false);
        acc11 = __builtin_amdgcn_wmma_f32_16x16x32_f16(false, a1, false, b1, (short)0, acc11, false, false);
    }

    float bv0 = bias ? bias[cb0] : 0.f;
    float bv1 = bias ? bias[cb1] : 0.f;
    int mr = m0 + 8 * (lane >> 4);
#pragma unroll
    for (int v = 0; v < 8; ++v) {
        int ma = mr + v;
        if (ma < M) {
            Y[(size_t)ma * N + cb0] = acc00[v] + bv0;
            Y[(size_t)ma * N + cb1] = acc01[v] + bv1;
        }
        int mb = mr + 16 + v;
        if (mb < M) {
            Y[(size_t)mb * N + cb0] = acc10[v] + bv0;
            Y[(size_t)mb * N + cb1] = acc11[v] + bv1;
        }
    }
}

// ---------------------------------------------------------------------------
// Edge pass 1: alpha[e,h] = <leaky_relu(xl[src]+xr[dst]), att[h]> ; seg-max.
// One wave per edge: lane = h*8 + ls, each lane covers C/8 channels.
// ---------------------------------------------------------------------------
__global__ void __launch_bounds__(256)
k_edge_alpha(const int* __restrict__ src, const int* __restrict__ dst,
             const float* __restrict__ xl, const float* __restrict__ xr,
             const float* __restrict__ att, float* __restrict__ alpha,
             unsigned* __restrict__ menc, int E, int C) {
    int wave = threadIdx.x >> 5, lane = threadIdx.x & 31;
    int e = blockIdx.x * 8 + wave;
    if (e >= E) return;
    int h = lane >> 3, ls = lane & 7;
    int s = src[e], d = dst[e];
    const float* pl = xl + ((size_t)s * NH + h) * C;
    const float* pr = xr + ((size_t)d * NH + h) * C;
    const float* pa = att + (size_t)h * C;
    float acc = 0.f;
    for (int c = ls; c < C; c += 8) {
        float g = pl[c] + pr[c];
        g = g > 0.f ? g : 0.2f * g;              // leaky_relu(0.2)
        acc += g * pa[c];
    }
    acc += __shfl_xor(acc, 1, 32);
    acc += __shfl_xor(acc, 2, 32);
    acc += __shfl_xor(acc, 4, 32);
    if (ls == 0) {
        alpha[(size_t)e * NH + h] = acc;
        unsigned u = __float_as_uint(acc);
        unsigned enc = (u & 0x80000000u) ? ~u : (u | 0x80000000u);  // order-preserving
        atomicMax(&menc[(size_t)d * NH + h], enc);
    }
}

// Edge pass 2: ea = exp(alpha - m[dst]); den[dst] += ea.  (alpha overwritten)
__global__ void k_edge_exp(const int* __restrict__ dst, float* __restrict__ alpha_ea,
                           const unsigned* __restrict__ menc, float* __restrict__ den,
                           int EH) {
    int i = blockIdx.x * blockDim.x + threadIdx.x;
    if (i >= EH) return;
    int e = i >> 2, h = i & 3;                   // NH == 4
    int d = dst[e];
    unsigned enc = menc[(size_t)d * NH + h];
    float m = (enc & 0x80000000u) ? __uint_as_float(enc ^ 0x80000000u)
                                  : __uint_as_float(~enc);
    float ea = expf(alpha_ea[i] - m);
    alpha_ea[i] = ea;
    atomicAdd(&den[(size_t)d * NH + h], ea);
}

// Edge pass 3: out[dst] += (ea/den) * xl[src]
__global__ void __launch_bounds__(256)
k_edge_scatter(const int* __restrict__ src, const int* __restrict__ dst,
               const float* __restrict__ xl, const float* __restrict__ ea,
               const float* __restrict__ den, float* __restrict__ out, int E, int C) {
    int wave = threadIdx.x >> 5, lane = threadIdx.x & 31;
    int e = blockIdx.x * 8 + wave;
    if (e >= E) return;
    int h = lane >> 3, ls = lane & 7;
    int s = src[e], d = dst[e];
    float coef = ea[(size_t)e * NH + h] / (den[(size_t)d * NH + h] + 1e-16f);
    const float* pl = xl + ((size_t)s * NH + h) * C;
    float*       po = out + ((size_t)d * NH + h) * C;
    for (int c = ls; c < C; c += 8) atomicAdd(&po[c], coef * pl[c]);
}

// ---------------------------------------------------------------------------
// host orchestration
// ---------------------------------------------------------------------------
extern "C" void kernel_launch(void* const* d_in, const int* in_sizes, int n_in,
                              void* d_out, int out_size, void* d_ws, size_t ws_size,
                              hipStream_t stream) {
    (void)in_sizes; (void)n_in; (void)out_size; (void)ws_size;

    const float* x_sample = (const float*)d_in[0];
    const float* x_gene   = (const float*)d_in[1];
    const int*   sg_src   = (const int*)d_in[2];
    const int*   sg_dst   = (const int*)d_in[3];
    const int*   gs_src   = (const int*)d_in[4];
    const int*   gs_dst   = (const int*)d_in[5];
    const float* Wl1_sg = (const float*)d_in[6];   const float* bl1_sg  = (const float*)d_in[7];
    const float* Wr1_sg = (const float*)d_in[8];   const float* br1_sg  = (const float*)d_in[9];
    const float* att1_sg = (const float*)d_in[10]; const float* bias1_sg = (const float*)d_in[11];
    const float* Wl1_gs = (const float*)d_in[12];  const float* bl1_gs  = (const float*)d_in[13];
    const float* Wr1_gs = (const float*)d_in[14];  const float* br1_gs  = (const float*)d_in[15];
    const float* att1_gs = (const float*)d_in[16]; const float* bias1_gs = (const float*)d_in[17];
    const float* Wl3 = (const float*)d_in[18];     const float* bl3 = (const float*)d_in[19];
    const float* Wr3 = (const float*)d_in[20];     const float* br3 = (const float*)d_in[21];
    const float* att3 = (const float*)d_in[22];    const float* bias3 = (const float*)d_in[23];
    const float* sl1_W = (const float*)d_in[24];   const float* sl1_b = (const float*)d_in[25];
    const float* sl3_W = (const float*)d_in[26];   const float* sl3_b = (const float*)d_in[27];

    char* p = (char*)d_ws;
    auto carve = [&](size_t bytes) -> char* {
        char* r = p; p += (bytes + 255) & ~(size_t)255; return r;
    };

    // f16 operand copies (weights transposed to [N,K])
    _Float16* hx_sample = (_Float16*)carve((size_t)NS * DIN * 2);
    _Float16* hx_gene   = (_Float16*)carve((size_t)NG * DIN * 2);
    _Float16* hWl1sg = (_Float16*)carve((size_t)DIN * HC1 * 2);
    _Float16* hWr1sg = (_Float16*)carve((size_t)DIN * HC1 * 2);
    _Float16* hWl1gs = (_Float16*)carve((size_t)DIN * HC1 * 2);
    _Float16* hWr1gs = (_Float16*)carve((size_t)DIN * HC1 * 2);
    _Float16* hWl3   = (_Float16*)carve((size_t)HC1 * HC3 * 2);
    _Float16* hWr3   = (_Float16*)carve((size_t)HC1 * HC3 * 2);
    _Float16* hsl1W  = (_Float16*)carve((size_t)DIN * C1 * 2);
    _Float16* hsl3W  = (_Float16*)carve((size_t)HC1 * C3 * 2);
    _Float16* hx1_gene   = (_Float16*)carve((size_t)NG * HC1 * 2);
    _Float16* hx1_sample = (_Float16*)carve((size_t)NS * HC1 * 2);
    // fp32 working buffers
    float* xl_sg = (float*)carve((size_t)NS * HC1 * 4);
    float* xr_sg = (float*)carve((size_t)NG * HC1 * 4);     // reused as xl_gs
    float* out1_gene   = (float*)carve((size_t)NG * HC1 * 4); // -> x1_gene (f32)
    float* xr_gs = (float*)carve((size_t)NS * HC1 * 4);
    float* out1_sample = (float*)carve((size_t)NS * HC1 * 4); // -> x1_sample (f32)
    float* sl1 = (float*)carve((size_t)NS * C1 * 4);
    float* xl3 = (float*)carve((size_t)NG * HC3 * 4);
    float* xr3 = (float*)carve((size_t)NS * HC3 * 4);
    float* acc3 = (float*)carve((size_t)NS * HC3 * 4);
    float* sl3 = (float*)carve((size_t)NS * C3 * 4);
    float* alpha = (float*)carve((size_t)NE * NH * 4);      // also holds ea after pass2
    unsigned* menc = (unsigned*)carve((size_t)NG * NH * 4);
    float* den = (float*)carve((size_t)NG * NH * 4);
    float* xl_gs = xr_sg;                                    // buffer reuse (stream-ordered)

    const int TB = 256;
#define EW(n) cdiv((n), TB), TB, 0, stream
    // --- f32 -> f16 conversions (activations direct, weights transposed) ---
    k_f32_to_f16<<<EW(NS * DIN)>>>(x_sample, hx_sample, NS * DIN);
    k_f32_to_f16<<<EW(NG * DIN)>>>(x_gene, hx_gene, NG * DIN);
    k_f32_to_f16_t<<<EW(DIN * HC1)>>>(Wl1_sg, hWl1sg, DIN, HC1);
    k_f32_to_f16_t<<<EW(DIN * HC1)>>>(Wr1_sg, hWr1sg, DIN, HC1);
    k_f32_to_f16_t<<<EW(DIN * HC1)>>>(Wl1_gs, hWl1gs, DIN, HC1);
    k_f32_to_f16_t<<<EW(DIN * HC1)>>>(Wr1_gs, hWr1gs, DIN, HC1);
    k_f32_to_f16_t<<<EW(HC1 * HC3)>>>(Wl3, hWl3, HC1, HC3);
    k_f32_to_f16_t<<<EW(HC1 * HC3)>>>(Wr3, hWr3, HC1, HC3);
    k_f32_to_f16_t<<<EW(DIN * C1)>>>(sl1_W, hsl1W, DIN, C1);
    k_f32_to_f16_t<<<EW(HC1 * C3)>>>(sl3_W, hsl3W, HC1, C3);

    auto gemm = [&](const _Float16* X, const _Float16* Wt, const float* b, float* Y,
                    int M, int N, int K) {
        dim3 g(cdiv(M, 128), cdiv(N, 64));
        k_gemm_wmma<<<g, 256, 0, stream>>>(X, Wt, b, Y, M, N, K);
    };

    // ================= layer 1, relation sg (sample -> gene) =================
    gemm(hx_sample, hWl1sg, bl1_sg, xl_sg, NS, HC1, DIN);
    gemm(hx_gene,   hWr1sg, br1_sg, xr_sg, NG, HC1, DIN);
    k_fill_u32<<<EW(NG * NH)>>>(menc, NEG_ENC, NG * NH);
    k_fill_f32<<<EW(NG * NH)>>>(den, 0.f, NG * NH);
    k_rowbias<<<EW(NG * HC1)>>>(out1_gene, bias1_sg, NG, HC1);
    k_edge_alpha<<<cdiv(NE, 8), 256, 0, stream>>>(sg_src, sg_dst, xl_sg, xr_sg, att1_sg,
                                                  alpha, menc, NE, C1);
    k_edge_exp<<<EW(NE * NH)>>>(sg_dst, alpha, menc, den, NE * NH);
    k_edge_scatter<<<cdiv(NE, 8), 256, 0, stream>>>(sg_src, sg_dst, xl_sg, alpha, den,
                                                    out1_gene, NE, C1);
    k_elu_f16copy<<<EW(NG * HC1)>>>(out1_gene, hx1_gene, NG * HC1);

    // ================= layer 1, relation gs (gene -> sample) =================
    gemm(hx_gene,   hWl1gs, bl1_gs, xl_gs, NG, HC1, DIN);
    gemm(hx_sample, hWr1gs, br1_gs, xr_gs, NS, HC1, DIN);
    gemm(hx_sample, hsl1W,  sl1_b,  sl1,   NS, C1,  DIN);
    k_fill_u32<<<EW(NS * NH)>>>(menc, NEG_ENC, NS * NH);
    k_fill_f32<<<EW(NS * NH)>>>(den, 0.f, NS * NH);
    k_rowbias<<<EW(NS * HC1)>>>(out1_sample, bias1_gs, NS, HC1);
    k_edge_alpha<<<cdiv(NE, 8), 256, 0, stream>>>(gs_src, gs_dst, xl_gs, xr_gs, att1_gs,
                                                  alpha, menc, NE, C1);
    k_edge_exp<<<EW(NE * NH)>>>(gs_dst, alpha, menc, den, NE * NH);
    k_edge_scatter<<<cdiv(NE, 8), 256, 0, stream>>>(gs_src, gs_dst, xl_gs, alpha, den,
                                                    out1_sample, NE, C1);
    k_sample1<<<EW(NS * HC1)>>>(out1_sample, sl1, hx1_sample, NS * HC1);

    // ================= layer 3, relation gs (gene -> sample) =================
    gemm(hx1_gene,   hWl3,  bl3,   xl3, NG, HC3, HC1);
    gemm(hx1_sample, hWr3,  br3,   xr3, NS, HC3, HC1);
    gemm(hx1_sample, hsl3W, sl3_b, sl3, NS, C3,  HC1);
    k_fill_u32<<<EW(NS * NH)>>>(menc, NEG_ENC, NS * NH);
    k_fill_f32<<<EW(NS * NH)>>>(den, 0.f, NS * NH);
    k_fill_f32<<<EW(NS * HC3)>>>(acc3, 0.f, NS * HC3);
    k_edge_alpha<<<cdiv(NE, 8), 256, 0, stream>>>(gs_src, gs_dst, xl3, xr3, att3,
                                                  alpha, menc, NE, C3);
    k_edge_exp<<<EW(NE * NH)>>>(gs_dst, alpha, menc, den, NE * NH);
    k_edge_scatter<<<cdiv(NE, 8), 256, 0, stream>>>(gs_src, gs_dst, xl3, alpha, den,
                                                    acc3, NE, C3);
    k_final<<<EW(NS * C3)>>>(acc3, sl3, bias3, (float*)d_out, NS * C3);
#undef EW
}
